// Attention_10325101379654
// MI455X (gfx1250) — matrix operand order repaired and checked by reading, not compiled, not run on hardware
//
#include <hip/hip_runtime.h>

// ---------------------------------------------------------------------------
// Causal MHA block for MI455X (gfx1250, wave32, WMMA + TDM).
//   D=1024, H=16, d_head=64, B=2, S=2048. fp32 in/out, f16 WMMA internally.
// Pipeline: cvt(f32->f16) -> QKV GEMM (V transposed) -> flash attention
//   (K tiles double-buffered via tensor_load_to_lds) -> output projection.
// GEMM waves compute 32x64 tiles: 12 b128 loads -> 8 WMMAs per K-step.
// ---------------------------------------------------------------------------

typedef __attribute__((ext_vector_type(16))) _Float16     v16h;
typedef __attribute__((ext_vector_type(8)))  _Float16     v8h;
typedef __attribute__((ext_vector_type(4)))  _Float16     v4h;
typedef __attribute__((ext_vector_type(8)))  float        v8f;
typedef __attribute__((ext_vector_type(4)))  unsigned int uint32x4;
typedef __attribute__((ext_vector_type(8)))  int          int32x8;
typedef __attribute__((ext_vector_type(4)))  int          int32x4;

union AFrag { v16h v; v8h h[2]; _Float16 e[16]; };
union CFrag { v8f  v; float e[8]; };

#define D_MODEL 1024
#define NH      16
#define DH      64
#define BB      2
#define SS      2048

static __device__ __forceinline__ v8f wmma_f16(const AFrag& a, const AFrag& b, v8f c) {
    return __builtin_amdgcn_wmma_f32_16x16x32_f16(false, a.v, false, b.v,
                                                  (short)0, c, false, false);
}

// TDM: DMA a [rows x 64] f16 tile (row-major, stride 64) from global to LDS.
static __device__ __forceinline__ void tdm_load_k_tile(const void* gptr,
                                                       unsigned lds_byte_off) {
    const unsigned long long gaddr = (unsigned long long)gptr;
    uint32x4 g0;
    g0[0] = 1u;                                   // count=1, user mode
    g0[1] = lds_byte_off;                         // lds_addr (bytes)
    g0[2] = (unsigned)gaddr;                      // global_addr[31:0]
    g0[3] = ((unsigned)(gaddr >> 32) & 0x01FFFFFFu) | 0x80000000u; // type=2
    int32x8 g1;
    g1[0] = 0x00010000;        // workgroup_mask=0, data_size=1 (2B)
    g1[1] = (int)(DH << 16);   // tensor_dim0 = 64
    g1[2] = (int)(SS << 16);   // tensor_dim1 = 2048
    g1[3] = (int)(DH << 16);   // tile_dim0 = 64
    g1[4] = 32;                // tile_dim1 = 32, tile_dim2 = 0
    g1[5] = DH;                // tensor_dim0_stride = 64
    g1[6] = 0;
    g1[7] = 0;
    const int32x4 gz = {0, 0, 0, 0};              // 2D: groups 2/3 unused
    asm volatile("tensor_load_to_lds %0, %1, %2, %3"
                 :: "s"(g0), "s"(g1), "s"(gz), "s"(gz)
                 : "memory");
}

// ---------------------------------------------------------------------------
// Kernel 0: one-time fp32 -> f16 conversion.
// ---------------------------------------------------------------------------
__global__ __launch_bounds__(256)
void cvt_f32_to_f16_kernel(const float* __restrict__ src,
                           _Float16* __restrict__ dst, int n4) {
    int i = blockIdx.x * 256 + threadIdx.x;
    const int stride = gridDim.x * 256;
    for (; i < n4; i += stride) {
        const float4 f = ((const float4*)src)[i];
        v4h h;
        h[0] = (_Float16)f.x; h[1] = (_Float16)f.y;
        h[2] = (_Float16)f.z; h[3] = (_Float16)f.w;
        ((v4h*)dst)[i] = h;
    }
}

// ---------------------------------------------------------------------------
// Kernel 1: Q/K/V projections, all-f16 operands. Wave computes 32x64 tile:
// load 2 A-frags + 4 B-frags, then 8 WMMAs. Q pre-scaled by 1/sqrt(DH).
// Q,K stored [b,h,s,dh]; V stored transposed [b,h,dh,s].
// ---------------------------------------------------------------------------
__global__ __launch_bounds__(256)
void qkv_gemm_kernel(const _Float16* __restrict__ xh,
                     const _Float16* __restrict__ wqh,
                     const _Float16* __restrict__ wkh,
                     const _Float16* __restrict__ wvh,
                     _Float16* __restrict__ qo,
                     _Float16* __restrict__ ko,
                     _Float16* __restrict__ vo) {
    const int wave   = threadIdx.x >> 5;
    const int lane   = threadIdx.x & 31;
    const int half16 = lane >> 4;
    const int l15    = lane & 15;
    const int nt64   = blockIdx.x * 8 + wave;   // 0..15
    const int mtile  = blockIdx.y;              // 0..127 (32-row tiles)
    const int wsel   = blockIdx.z;              // 0=Q, 1=K, 2=V

    const _Float16* W = (wsel == 0) ? wqh : (wsel == 1) ? wkh : wvh;
    const float scale = (wsel == 0) ? 0.125f : 1.0f;

    const int m0 = mtile * 32, n0 = nt64 * 64;
    const _Float16* arow0 = xh + (size_t)(m0 + l15) * D_MODEL + half16 * 8;
    const _Float16* arow1 = arow0 + (size_t)16 * D_MODEL;
    const _Float16* brow0 = W + (size_t)(n0 + l15) * D_MODEL + half16 * 8;

    CFrag acc[2][4];
#pragma unroll
    for (int r = 0; r < 2; ++r)
#pragma unroll
        for (int t = 0; t < 4; ++t) acc[r][t].v = (v8f){};

    for (int kk = 0; kk < D_MODEL; kk += 32) {
        AFrag a0, a1, b[4];
        a0.h[0] = *(const v8h*)(arow0 + kk);
        a0.h[1] = *(const v8h*)(arow0 + kk + 16);
        a1.h[0] = *(const v8h*)(arow1 + kk);
        a1.h[1] = *(const v8h*)(arow1 + kk + 16);
#pragma unroll
        for (int t = 0; t < 4; ++t) {
            const _Float16* bp = brow0 + (size_t)t * 16 * D_MODEL + kk;
            b[t].h[0] = *(const v8h*)(bp);
            b[t].h[1] = *(const v8h*)(bp + 16);
        }
        __builtin_prefetch(arow0 + kk + 32, 0, 1);
        __builtin_prefetch(arow1 + kk + 32, 0, 1);
#pragma unroll
        for (int t = 0; t < 4; ++t) {
            acc[0][t].v = wmma_f16(a0, b[t], acc[0][t].v);
            acc[1][t].v = wmma_f16(a1, b[t], acc[1][t].v);
        }
    }

    const int h = nt64;
    if (wsel == 2) {      // V transposed: [b,h,dh,s]
#pragma unroll
        for (int r = 0; r < 2; ++r)
#pragma unroll
            for (int t = 0; t < 4; ++t) {
                const int dh = t * 16 + l15;
                for (int i = 0; i < 8; ++i) {
                    const int m  = m0 + r * 16 + half16 * 8 + i;
                    const int bi = m >> 11, si = m & (SS - 1);
                    vo[(((size_t)bi * NH + h) * DH + dh) * SS + si] =
                        (_Float16)acc[r][t].e[i];
                }
            }
    } else {              // Q/K: [b,h,s,dh]
        _Float16* out = (wsel == 0) ? qo : ko;
#pragma unroll
        for (int r = 0; r < 2; ++r)
#pragma unroll
            for (int t = 0; t < 4; ++t) {
                const int dh = t * 16 + l15;
                for (int i = 0; i < 8; ++i) {
                    const int m  = m0 + r * 16 + half16 * 8 + i;
                    const int bi = m >> 11, si = m & (SS - 1);
                    out[(((size_t)bi * NH + h) * SS + si) * DH + dh] =
                        (_Float16)(acc[r][t].e[i] * scale);
                }
            }
    }
}

// ---------------------------------------------------------------------------
// Kernel 2: causal flash attention, one wave per (b,h, 16-query tile).
// K tiles (32x64 f16) double-buffered into LDS via TDM: issue chunk i+1,
// s_wait_tensorcnt 1, compute chunk i. 4 waves/block, 36 KB LDS.
// ---------------------------------------------------------------------------
__global__ __launch_bounds__(128)
void flash_attn_kernel(const _Float16* __restrict__ q,
                       const _Float16* __restrict__ k,
                       const _Float16* __restrict__ vT,
                       _Float16* __restrict__ attn) {
    // per wave: 2 K-stage buffers (2 x 4KB) then P tile (1KB)
    __shared__ __align__(16) char smem[4 * 8192 + 4 * 1024];
    const int wave   = threadIdx.x >> 5;
    const int lane   = threadIdx.x & 31;
    const int half16 = lane >> 4;
    const int l15    = lane & 15;
    const int qt     = blockIdx.x * 4 + wave;  // 0..127
    const int bh     = blockIdx.y;             // 0..31
    const size_t base = (size_t)bh * SS * DH;
    const _Float16* qb  = q  + base;
    const _Float16* kbp = k  + base;
    const _Float16* vb  = vT + base;           // [dh][s]
    const int qbase = qt * 16;

    const unsigned kst_base = (unsigned)(wave * 8192);
    _Float16* pt = (_Float16*)(smem + 4 * 8192 + wave * 1024);

    // Q fragments (A-layout), contraction over dh split [0,32) / [32,64)
    AFrag aq0, aq1;
    {
        const _Float16* qp = qb + (size_t)(qbase + l15) * DH + half16 * 8;
        aq0.h[0] = *(const v8h*)(qp);
        aq0.h[1] = *(const v8h*)(qp + 16);
        aq1.h[0] = *(const v8h*)(qp + 32);
        aq1.h[1] = *(const v8h*)(qp + 48);
    }

    CFrag o0, o1, o2, o3;
    o0.v = (v8f){}; o1.v = (v8f){}; o2.v = (v8f){}; o3.v = (v8f){};
    CFrag mrow, lrow;
    for (int i = 0; i < 8; ++i) { mrow.e[i] = -1e30f; lrow.e[i] = 0.0f; }

    const int nchunks = (qbase + 16 + 31) / 32;
    // prime the pipeline: chunk 0 -> buffer 0
    tdm_load_k_tile(kbp, kst_base);

    for (int c = 0; c < nchunks; ++c) {
        const int kb0 = c * 32;
        const int cur = c & 1;
        if (c + 1 < nchunks) {   // issue next chunk into the other buffer
            tdm_load_k_tile(kbp + (size_t)(kb0 + 32) * DH,
                            kst_base + (unsigned)((cur ^ 1) * 4096));
            asm volatile("s_wait_tensorcnt 0x1" ::: "memory");
        } else {
            asm volatile("s_wait_tensorcnt 0x0" ::: "memory");
        }
        const _Float16* kst = (const _Float16*)(smem + kst_base + cur * 4096);

        // K fragments from LDS (B-layout: K=dh contraction, N=key)
        AFrag bk00, bk01, bk10, bk11;
        {
            const _Float16* kp0 = kst + l15 * DH + half16 * 8;
            bk00.h[0] = *(const v8h*)(kp0);      bk00.h[1] = *(const v8h*)(kp0 + 16);
            bk01.h[0] = *(const v8h*)(kp0 + 32); bk01.h[1] = *(const v8h*)(kp0 + 48);
            const _Float16* kp1 = kp0 + 16 * DH;
            bk10.h[0] = *(const v8h*)(kp1);      bk10.h[1] = *(const v8h*)(kp1 + 16);
            bk11.h[0] = *(const v8h*)(kp1 + 32); bk11.h[1] = *(const v8h*)(kp1 + 48);
        }
        CFrag s0, s1; s0.v = (v8f){}; s1.v = (v8f){};
        s0.v = wmma_f16(aq0, bk00, s0.v);
        s0.v = wmma_f16(aq1, bk01, s0.v);
        s1.v = wmma_f16(aq0, bk10, s1.v);
        s1.v = wmma_f16(aq1, bk11, s1.v);

        // causal mask + online softmax (row reductions within 16-lane halves)
        const int kc0 = kb0 + l15, kc1 = kc0 + 16;
        for (int i = 0; i < 8; ++i) {
            const int qrow = qbase + half16 * 8 + i;
            if (kc0 > qrow) s0.e[i] = -1e30f;
            if (kc1 > qrow) s1.e[i] = -1e30f;
            float mx = fmaxf(s0.e[i], s1.e[i]);
            for (int d = 1; d < 16; d <<= 1)
                mx = fmaxf(mx, __shfl_xor(mx, d, 32));
            const float mnew = fmaxf(mrow.e[i], mx);
            const float csc  = __expf(mrow.e[i] - mnew);
            mrow.e[i] = mnew;
            const float p0 = __expf(s0.e[i] - mnew);
            const float p1 = __expf(s1.e[i] - mnew);
            float ps = p0 + p1;
            for (int d = 1; d < 16; d <<= 1)
                ps += __shfl_xor(ps, d, 32);
            lrow.e[i] = lrow.e[i] * csc + ps;
            o0.e[i] *= csc; o1.e[i] *= csc; o2.e[i] *= csc; o3.e[i] *= csc;
            const int r = half16 * 8 + i;
            pt[r * 32 + l15]      = (_Float16)p0;
            pt[r * 32 + l15 + 16] = (_Float16)p1;
        }
        asm volatile("s_wait_dscnt 0x0" ::: "memory");

        // P fragment (A-layout 16x32) from LDS
        AFrag ap;
        {
            const _Float16* lp = pt + l15 * 32 + half16 * 8;
            ap.h[0] = *(const v8h*)(lp);
            ap.h[1] = *(const v8h*)(lp + 16);
        }
        // V fragments from transposed V: contiguous along keys
    const size_t vcol = (size_t)kb0 + half16 * 8;
#pragma unroll
        for (int dhc = 0; dhc < 4; ++dhc) {
            AFrag bv;
            const _Float16* vrow = vb + (size_t)(dhc * 16 + l15) * SS + vcol;
            bv.h[0] = *(const v8h*)(vrow);
            bv.h[1] = *(const v8h*)(vrow + 16);
            CFrag* oc = (dhc == 0) ? &o0 : (dhc == 1) ? &o1 : (dhc == 2) ? &o2 : &o3;
            oc->v = wmma_f16(ap, bv, oc->v);
        }
    }

    // normalize and emit attn [b, s, D] f16 for the projection GEMM
    const int bi = bh >> 4, h = bh & 15;
    for (int i = 0; i < 8; ++i) {
        const float inv  = 1.0f / lrow.e[i];
        const int   srow = qbase + half16 * 8 + i;
        _Float16* op = attn + ((size_t)bi * SS + srow) * D_MODEL + h * DH + l15;
        op[0]  = (_Float16)(o0.e[i] * inv);
        op[16] = (_Float16)(o1.e[i] * inv);
        op[32] = (_Float16)(o2.e[i] * inv);
        op[48] = (_Float16)(o3.e[i] * inv);
    }
}

// ---------------------------------------------------------------------------
// Kernel 3: output projection, all-f16 operands, 32x64 tile per wave, f32 out.
// ---------------------------------------------------------------------------
__global__ __launch_bounds__(256)
void proj_gemm_kernel(const _Float16* __restrict__ attn,
                      const _Float16* __restrict__ wph,
                      float* __restrict__ out) {
    const int wave   = threadIdx.x >> 5;
    const int lane   = threadIdx.x & 31;
    const int half16 = lane >> 4;
    const int l15    = lane & 15;
    const int nt64   = blockIdx.x * 8 + wave;  // 0..15
    const int mtile  = blockIdx.y;             // 0..127
    const int m0 = mtile * 32, n0 = nt64 * 64;

    const _Float16* arow0 = attn + (size_t)(m0 + l15) * D_MODEL + half16 * 8;
    const _Float16* arow1 = arow0 + (size_t)16 * D_MODEL;
    const _Float16* brow0 = wph  + (size_t)(n0 + l15) * D_MODEL + half16 * 8;

    CFrag acc[2][4];
#pragma unroll
    for (int r = 0; r < 2; ++r)
#pragma unroll
        for (int t = 0; t < 4; ++t) acc[r][t].v = (v8f){};

    for (int kk = 0; kk < D_MODEL; kk += 32) {
        AFrag a0, a1, b[4];
        a0.h[0] = *(const v8h*)(arow0 + kk);
        a0.h[1] = *(const v8h*)(arow0 + kk + 16);
        a1.h[0] = *(const v8h*)(arow1 + kk);
        a1.h[1] = *(const v8h*)(arow1 + kk + 16);
#pragma unroll
        for (int t = 0; t < 4; ++t) {
            const _Float16* bp = brow0 + (size_t)t * 16 * D_MODEL + kk;
            b[t].h[0] = *(const v8h*)(bp);
            b[t].h[1] = *(const v8h*)(bp + 16);
        }
        __builtin_prefetch(arow0 + kk + 32, 0, 1);
        __builtin_prefetch(arow1 + kk + 32, 0, 1);
#pragma unroll
        for (int t = 0; t < 4; ++t) {
            acc[0][t].v = wmma_f16(a0, b[t], acc[0][t].v);
            acc[1][t].v = wmma_f16(a1, b[t], acc[1][t].v);
        }
    }
#pragma unroll
    for (int r = 0; r < 2; ++r)
#pragma unroll
        for (int t = 0; t < 4; ++t) {
            const int n = n0 + t * 16 + l15;
            for (int i = 0; i < 8; ++i) {
                const int m = m0 + r * 16 + half16 * 8 + i;
                out[(size_t)m * D_MODEL + n] = acc[r][t].e[i];
            }
        }
}

// ---------------------------------------------------------------------------
extern "C" void kernel_launch(void* const* d_in, const int* in_sizes, int n_in,
                              void* d_out, int out_size, void* d_ws, size_t ws_size,
                              hipStream_t stream) {
    const float* x  = (const float*)d_in[0];
    const float* Wq = (const float*)d_in[1];
    const float* Wk = (const float*)d_in[2];
    const float* Wv = (const float*)d_in[3];
    const float* Wp = (const float*)d_in[4];
    float* out = (float*)d_out;

    const size_t N1 = (size_t)BB * NH * SS * DH;       // 4,194,304
    const size_t W2 = (size_t)D_MODEL * D_MODEL;       // 1,048,576
    _Float16* qw  = (_Float16*)d_ws;
    _Float16* kw  = qw  + N1;
    _Float16* vw  = kw  + N1;   // transposed V
    _Float16* aw  = vw  + N1;
    _Float16* xh  = aw  + N1;
    _Float16* wqh = xh  + N1;
    _Float16* wkh = wqh + W2;
    _Float16* wvh = wkh + W2;
    _Float16* wph = wvh + W2;

    dim3 blk(256);
    cvt_f32_to_f16_kernel<<<512, blk, 0, stream>>>(x,  xh,  (int)(N1 / 4));
    cvt_f32_to_f16_kernel<<<256, blk, 0, stream>>>(Wq, wqh, (int)(W2 / 4));
    cvt_f32_to_f16_kernel<<<256, blk, 0, stream>>>(Wk, wkh, (int)(W2 / 4));
    cvt_f32_to_f16_kernel<<<256, blk, 0, stream>>>(Wv, wvh, (int)(W2 / 4));
    cvt_f32_to_f16_kernel<<<256, blk, 0, stream>>>(Wp, wph, (int)(W2 / 4));

    // 32-row M tiles: B*S/32 = 128; 64-wide N tiles: D/64 = 16
    qkv_gemm_kernel<<<dim3(2, 128, 3), blk, 0, stream>>>(xh, wqh, wkh, wvh, qw, kw, vw);
    flash_attn_kernel<<<dim3(32, 32), dim3(128), 0, stream>>>(qw, kw, vw, aw);
    proj_gemm_kernel<<<dim3(2, 128), blk, 0, stream>>>(aw, wph, out);
}
